// ReinforceDistributed_22247930593334
// MI455X (gfx1250) — compile-verified, matrix operand order checked by reading, not run on hardware
//
#include <hip/hip_runtime.h>
#include <hip/hip_bf16.h>

typedef __attribute__((ext_vector_type(16))) _Float16 v16h;
typedef __attribute__((ext_vector_type(8)))  float    v8f;

#define HALL_SZ    4096
#define NSPLIT     3
#define EPS_DRAWS  10
#define NTHREADS   256
#define NWAVES     (NTHREADS / 32)
#define TILE_ITEMS 32
#define NTILES     (HALL_SZ / TILE_ITEMS)
#define ITEMS_PER_THREAD (HALL_SZ / NTHREADS)   // 16

__device__ __forceinline__ unsigned rotl32(unsigned x, int r) {
  return (x << r) | (x >> (32 - r));
}

// threefry2x32-20, returns x0. key=(k0,k1), counter=(c0,c1).
__device__ __forceinline__ unsigned threefry2x32_hi(unsigned k0, unsigned k1,
                                                    unsigned c0, unsigned c1) {
  const unsigned ks2 = k0 ^ k1 ^ 0x1BD11BDAu;
  unsigned ks[3] = {k0, k1, ks2};
  unsigned x0 = c0 + k0, x1 = c1 + k1;
  const int RA[4] = {13, 15, 26, 6};
  const int RB[4] = {17, 29, 16, 24};
  #pragma unroll
  for (int i = 0; i < 5; ++i) {
    #pragma unroll
    for (int r = 0; r < 4; ++r) {
      const int rot = (i & 1) ? RB[r] : RA[r];
      x0 += x1; x1 = rotl32(x1, rot); x1 ^= x0;
    }
    x0 += ks[(i + 1) % 3];
    x1 += ks[(i + 2) % 3] + (unsigned)(i + 1);
  }
  return x0;
}

__global__ __launch_bounds__(NTHREADS)
void reinforce_sample_kernel(const float* __restrict__ X,
                             const float* __restrict__ W,
                             const float* __restrict__ B,
                             int* __restrict__ out)
{
  __shared__ float s_logit[HALL_SZ];   // logits, then overwritten with probs
  __shared__ float s_cum[HALL_SZ];     // inclusive CDF
  __shared__ float s_red[NTHREADS];
  __shared__ int   s_redi[NTHREADS];
  __shared__ int   s_samples[EPS_DRAWS];

  const int row  = blockIdx.x;
  const int tid  = threadIdx.x;
  const int lane = tid & 31;
  const int wave = tid >> 5;
  const float* __restrict__ xrow = X + (size_t)row * (HALL_SZ * NSPLIT);

  // Preload per-split weights (uniform) and convert to f16 for WMMA A/B.
  _Float16 wh[NSPLIT][3];
  float    bb[NSPLIT];
  #pragma unroll
  for (int d = 0; d < NSPLIT; ++d) {
    wh[d][0] = (_Float16)W[d * 3 + 0];
    wh[d][1] = (_Float16)W[d * 3 + 1];
    wh[d][2] = (_Float16)W[d * 3 + 2];
    bb[d]    = B[d];
  }

  // ---- Phase 1: logits via V_WMMA_F32_16X16X32_F16 (K=3 of 32 used) ----
  for (int tile = wave; tile < NTILES; tile += NWAVES) {
    const int item0 = tile * TILE_ITEMS;
    const int d = (item0 < 2048) ? 0 : ((item0 < 3072) ? 1 : 2);

    // 32 lanes load 32 contiguous items' (3 floats each): 384B/wave, coalesced.
    const float* p = xrow + (size_t)(item0 + lane) * 3;
    const float f0 = p[0], f1 = p[1], f2 = p[2];

    if (tile + NWAVES < NTILES)
      __builtin_prefetch(xrow + (size_t)((tile + NWAVES) * TILE_ITEMS + lane) * 3, 0, 1);

    const bool lo = (lane < 16);
    const _Float16 z = (_Float16)0.f;

    // B-matrix 32x16: lanes 0-15 hold K=0..7 and K=16..23 (two per VGPR); only K<3 nonzero,
    // every column = W[d] so every D column equals the row's logit.
    v16h bmat = {};
    bmat[0] = lo ? wh[d][0] : z;
    bmat[1] = lo ? wh[d][1] : z;
    bmat[2] = lo ? wh[d][2] : z;

    // A-matrix 16x32: lane m (0-15) = row m, halves 0..3 = K0..3.
    v16h a1 = {};
    a1[0] = lo ? (_Float16)f0 : z;
    a1[1] = lo ? (_Float16)f1 : z;
    a1[2] = lo ? (_Float16)f2 : z;
    v8f c1 = {};
    c1 = __builtin_amdgcn_wmma_f32_16x16x32_f16(false, a1, false, bmat,
                                                (short)0, c1, false, false);

    // Second 16 items live in lanes 16-31: shuffle them down into lanes 0-15.
    const int src = (lane & 15) + 16;
    const float g0 = __shfl(f0, src, 32);
    const float g1 = __shfl(f1, src, 32);
    const float g2 = __shfl(f2, src, 32);
    v16h a2 = {};
    a2[0] = lo ? (_Float16)g0 : z;
    a2[1] = lo ? (_Float16)g1 : z;
    a2[2] = lo ? (_Float16)g2 : z;
    v8f c2 = {};
    c2 = __builtin_amdgcn_wmma_f32_16x16x32_f16(false, a2, false, bmat,
                                                (short)0, c2, false, false);

    // D layout: lane<16 VGPR j = M=j; lane>=16 VGPR j = M=j+8 (columns identical).
    if ((lane & 15) == 0) {
      const int mofs = (lane >> 4) * 8;   // 0 or 8
      #pragma unroll
      for (int j = 0; j < 8; ++j) {
        s_logit[item0 + mofs + j]      = c1[j] + bb[d];
        s_logit[item0 + 16 + mofs + j] = c2[j] + bb[d];
      }
    }
  }
  __syncthreads();

  // ---- Phase 2: per-split softmax + global argmax of probs ----
  const int sstart[NSPLIT] = {0, 2048, 3072};
  const int send[NSPLIT]   = {2048, 3072, 4096};
  float bestp = -1.f;
  int   besti = 0;

  for (int d = 0; d < NSPLIT; ++d) {
    // argmax of logits within split (first occurrence on ties)
    float lm = -INFINITY; int li = send[d];
    for (int i = sstart[d] + tid; i < send[d]; i += NTHREADS) {
      const float v = s_logit[i];
      if (v > lm) { lm = v; li = i; }
    }
    s_red[tid] = lm; s_redi[tid] = li;
    __syncthreads();
    for (int s = NTHREADS / 2; s > 0; s >>= 1) {
      if (tid < s) {
        const float ov = s_red[tid + s]; const int oi = s_redi[tid + s];
        if (ov > s_red[tid] || (ov == s_red[tid] && oi < s_redi[tid])) {
          s_red[tid] = ov; s_redi[tid] = oi;
        }
      }
      __syncthreads();
    }
    const float mx = s_red[0];
    const int   mi = s_redi[0];
    __syncthreads();

    // sum of exp(logit - max)
    float ls = 0.f;
    for (int i = sstart[d] + tid; i < send[d]; i += NTHREADS)
      ls += __expf(s_logit[i] - mx);
    s_red[tid] = ls;
    __syncthreads();
    for (int s = NTHREADS / 2; s > 0; s >>= 1) {
      if (tid < s) s_red[tid] += s_red[tid + s];
      __syncthreads();
    }
    const float logZ = mx + __logf(s_red[0]);

    // overwrite logits with probs for this split
    for (int i = sstart[d] + tid; i < send[d]; i += NTHREADS)
      s_logit[i] = __expf(s_logit[i] - logZ);

    const float bp = __expf(mx - logZ);    // max prob of this split
    if (bp > bestp) { bestp = bp; besti = mi; }
    __syncthreads();
  }

  // ---- Phase 3: inclusive CDF over all 4096 probs (total ~= 3.0) ----
  const int base = tid * ITEMS_PER_THREAD;
  float run = 0.f;
  #pragma unroll
  for (int j = 0; j < ITEMS_PER_THREAD; ++j) {
    run += s_logit[base + j];
    s_cum[base + j] = run;
  }
  s_red[tid] = run;
  __syncthreads();
  for (int off = 1; off < NTHREADS; off <<= 1) {   // Hillis-Steele inclusive scan
    const float v = (tid >= off) ? s_red[tid - off] : 0.f;
    __syncthreads();
    s_red[tid] += v;
    __syncthreads();
  }
  const float prefix = s_red[tid] - run;           // exclusive prefix
  const float total  = s_red[NTHREADS - 1];
  #pragma unroll
  for (int j = 0; j < ITEMS_PER_THREAD; ++j)
    s_cum[base + j] += prefix;
  __syncthreads();

  // ---- Phase 4: EPSILON inverse-CDF draws (counter-based PRNG, key 42) ----
  if (tid < EPS_DRAWS) {
    const unsigned r = threefry2x32_hi(0u, 42u, (unsigned)row, (unsigned)tid);
    const float u = (float)(r >> 8) * (1.f / 16777216.f);   // [0,1)
    const float target = u * total;
    int lo2 = 0, hi2 = HALL_SZ - 1;
    while (lo2 < hi2) {
      const int mid = (lo2 + hi2) >> 1;
      if (s_cum[mid] > target) hi2 = mid; else lo2 = mid + 1;
    }
    s_samples[tid] = lo2;
  }
  __syncthreads();

  if (tid == 0) {
    int hit = 0;
    #pragma unroll
    for (int e = 0; e < EPS_DRAWS; ++e) hit |= (s_samples[e] == besti);
    out[row] = hit ? besti : s_samples[EPS_DRAWS - 1];
  }
}

extern "C" void kernel_launch(void* const* d_in, const int* in_sizes, int n_in,
                              void* d_out, int out_size, void* d_ws, size_t ws_size,
                              hipStream_t stream) {
  const float* X = (const float*)d_in[0];   // (8192, 12288)
  const float* W = (const float*)d_in[1];   // (3, 3)
  const float* B = (const float*)d_in[2];   // (3,)
  int* out = (int*)d_out;                   // (8192,) int32 actions
  const int rows = out_size;                // 8192
  reinforce_sample_kernel<<<rows, NTHREADS, 0, stream>>>(X, W, B, out);
}
